// MultiheadSelfAttention_87016037417425
// MI455X (gfx1250) — compile-verified, matrix operand order checked by reading, not run on hardware
//
#include <hip/hip_runtime.h>
#include <hip/hip_bf16.h>

typedef __attribute__((ext_vector_type(16))) _Float16 v16h;
typedef __attribute__((ext_vector_type(8)))  _Float16 v8h;
typedef __attribute__((ext_vector_type(4)))  _Float16 v4h;
typedef __attribute__((ext_vector_type(8)))  float    v8f;

#define DM    1024
#define NH    16
#define HD    64
#define BB    2
#define SS    2048
#define MROWS (BB * SS)   // 4096

#define LDSTRIDE 40       // halves per tile row (80B: 16B-aligned, conflict-free)

// ---------------------------------------------------------------------------
// A-fragment loader (global) for V_WMMA_F32_16X16X32_F16 (ISA 7.12.2):
//   lanes 0-15 : M=lane,    halves 0..7 = K 0..7,  halves 8..15 = K 16..23
//   lanes 16-31: M=lane-16, halves 0..7 = K 8..15, halves 8..15 = K 24..31
// ---------------------------------------------------------------------------
__device__ __forceinline__ v16h frag_a16(const _Float16* __restrict__ p, int ld) {
  const int l  = threadIdx.x & 31;
  const int r  = l & 15;
  const int kb = (l >> 4) << 3;
  const _Float16* q = p + (size_t)r * ld + kb;
  v8h lo = *(const v8h*)(q);
  v8h hi = *(const v8h*)(q + 16);
  v16h out;
#pragma unroll
  for (int i = 0; i < 8; ++i) { out[i] = lo[i]; out[i + 8] = hi[i]; }
  return out;
}

__device__ __forceinline__ v8f wmma16(v16h a, v16h b, v8f c) {
  return __builtin_amdgcn_wmma_f32_16x16x32_f16(false, a, false, b, (short)0, c,
                                                false, false);
}

// async copy of one 16B chunk into LDS (ASYNCcnt-tracked)
__device__ __forceinline__ void async_b128(unsigned lds_addr, const _Float16* gp) {
  asm volatile("global_load_async_to_lds_b128 %0, %1, off"
               :: "v"(lds_addr), "v"(gp) : "memory");
}

// ---------------------------------------------------------------------------
// fp32 -> f16 cast
// ---------------------------------------------------------------------------
__global__ void cast_f16_kernel(const float* __restrict__ in,
                                _Float16* __restrict__ out, int n4) {
  const int i = blockIdx.x * blockDim.x + threadIdx.x;
  if (i < n4) {
    const float4 f = ((const float4*)in)[i];
    v4h o;
    o[0] = (_Float16)f.x; o[1] = (_Float16)f.y;
    o[2] = (_Float16)f.z; o[3] = (_Float16)f.w;
    ((v4h*)out)[i] = o;
  }
}

// ===========================================================================
// LDS-staged GEMM core macro-structure (shared by QKV and O kernels):
//   block tile 128(M) x 128(N), K-steps of 32, double-buffered async staging.
//   8 waves as 2(M) x 4(N); each wave computes 64x32 via 4x2 accumulators.
// ===========================================================================

// QKV projection: C[m,n] = sum_k X[m,k] * W[n,k]   (x @ W^T)
// grid = (M/128, N/128, 3); z=0 -> Q [b,h,s,d], z=1 -> K, z=2 -> V^T [b,h,d,s]
__global__ void __launch_bounds__(256)
qkv_gemm_kernel(const _Float16* __restrict__ X,
                const _Float16* __restrict__ Wq,
                const _Float16* __restrict__ Wk,
                const _Float16* __restrict__ Wv,
                _Float16* __restrict__ Qo,
                _Float16* __restrict__ Ko,
                _Float16* __restrict__ Vt) {
  __shared__ _Float16 sA[2][128 * LDSTRIDE];
  __shared__ _Float16 sB[2][128 * LDSTRIDE];

  const int z = blockIdx.z;
  const _Float16* W = (z == 0) ? Wq : (z == 1) ? Wk : Wv;
  const int mB = blockIdx.x * 128;
  const int nB = blockIdx.y * 128;
  const int w  = threadIdx.x >> 5;
  const int l  = threadIdx.x & 31;
  const int wm = (w >> 2) * 64;      // wave M offset in block
  const int wn = (w & 3) * 32;       // wave N offset in block
  const int lr = l & 15;
  const int kb = (l >> 4) << 3;

  // stage one 128x32 tile of A and of B: 512 chunks each, 2 per thread
  auto issue_tile = [&](int buf, int k0) {
#pragma unroll
    for (int i = 0; i < 2; ++i) {
      const int c   = threadIdx.x * 2 + i;   // 0..511
      const int row = c >> 2;                // 0..127
      const int off = (c & 3) * 8;           // half offset: 0,8,16,24
      async_b128((unsigned)(size_t)&sA[buf][row * LDSTRIDE + off],
                 X + (size_t)(mB + row) * DM + k0 + off);
      async_b128((unsigned)(size_t)&sB[buf][row * LDSTRIDE + off],
                 W + (size_t)(nB + row) * DM + k0 + off);
    }
  };

  auto frag_lds = [&](const _Float16* base, int row0) -> v16h {
    const _Float16* q = base + (row0 + lr) * LDSTRIDE + kb;
    v8h lo = *(const v8h*)(q);
    v8h hi = *(const v8h*)(q + 16);
    v16h out;
#pragma unroll
    for (int i = 0; i < 8; ++i) { out[i] = lo[i]; out[i + 8] = hi[i]; }
    return out;
  };

  v8f acc[4][2] = {};

  issue_tile(0, 0);
  for (int k0 = 0, step = 0; k0 < DM; k0 += 32, ++step) {
    const int  cur  = step & 1;
    const bool more = (k0 + 32) < DM;
    if (more) {
      issue_tile(cur ^ 1, k0 + 32);
      asm volatile("s_wait_asynccnt 0x4" ::: "memory");
    } else {
      asm volatile("s_wait_asynccnt 0x0" ::: "memory");
    }
    __syncthreads();

    v16h af[4], bf[2];
#pragma unroll
    for (int mt = 0; mt < 4; ++mt) af[mt] = frag_lds(sA[cur], wm + mt * 16);
#pragma unroll
    for (int nt = 0; nt < 2; ++nt) bf[nt] = frag_lds(sB[cur], wn + nt * 16);
#pragma unroll
    for (int mt = 0; mt < 4; ++mt)
#pragma unroll
      for (int nt = 0; nt < 2; ++nt)
        acc[mt][nt] = wmma16(af[mt], bf[nt], acc[mt][nt]);

    __syncthreads();   // readers done before next issue overwrites this buffer
  }

  const int col   = l & 15;           // C-frag: N = lane&15
  const int rbase = (l >> 4) << 3;    // C-frag: M = v + 8*(lane>=16)

  if (z < 2) {
    _Float16* O = (z == 0) ? Qo : Ko;
#pragma unroll
    for (int mt = 0; mt < 4; ++mt)
#pragma unroll
      for (int nt = 0; nt < 2; ++nt) {
        const int n = nB + wn + nt * 16 + col;
        const int h = n >> 6, d = n & (HD - 1);
#pragma unroll
        for (int v = 0; v < 8; ++v) {
          const int m = mB + wm + mt * 16 + rbase + v;
          const int b = m >> 11, s = m & (SS - 1);
          O[((size_t)((b * NH + h) * SS + s)) * HD + d] =
              (_Float16)acc[mt][nt][v];
        }
      }
  } else {
    // V transposed: consecutive C rows (s) per lane -> packed b128 stores
#pragma unroll
    for (int mt = 0; mt < 4; ++mt)
#pragma unroll
      for (int nt = 0; nt < 2; ++nt) {
        const int n = nB + wn + nt * 16 + col;
        const int h = n >> 6, d = n & (HD - 1);
        const int m = mB + wm + mt * 16 + rbase;
        const int b = m >> 11, s = m & (SS - 1);
        v8h pk;
#pragma unroll
        for (int v = 0; v < 8; ++v) pk[v] = (_Float16)acc[mt][nt][v];
        *(v8h*)(Vt + ((size_t)((b * NH + h) * HD + d)) * SS + s) = pk;
      }
  }
}

// Output projection: Out[m,n] = sum_k Attn[m,k] * Wo[n,k], fp32 result.
__global__ void __launch_bounds__(256)
o_gemm_kernel(const _Float16* __restrict__ Ain,
              const _Float16* __restrict__ Wo,
              float* __restrict__ Out) {
  __shared__ _Float16 sA[2][128 * LDSTRIDE];
  __shared__ _Float16 sB[2][128 * LDSTRIDE];

  const int mB = blockIdx.x * 128;
  const int nB = blockIdx.y * 128;
  const int w  = threadIdx.x >> 5;
  const int l  = threadIdx.x & 31;
  const int wm = (w >> 2) * 64;
  const int wn = (w & 3) * 32;
  const int lr = l & 15;
  const int kb = (l >> 4) << 3;

  auto issue_tile = [&](int buf, int k0) {
#pragma unroll
    for (int i = 0; i < 2; ++i) {
      const int c   = threadIdx.x * 2 + i;
      const int row = c >> 2;
      const int off = (c & 3) * 8;
      async_b128((unsigned)(size_t)&sA[buf][row * LDSTRIDE + off],
                 Ain + (size_t)(mB + row) * DM + k0 + off);
      async_b128((unsigned)(size_t)&sB[buf][row * LDSTRIDE + off],
                 Wo + (size_t)(nB + row) * DM + k0 + off);
    }
  };

  auto frag_lds = [&](const _Float16* base, int row0) -> v16h {
    const _Float16* q = base + (row0 + lr) * LDSTRIDE + kb;
    v8h lo = *(const v8h*)(q);
    v8h hi = *(const v8h*)(q + 16);
    v16h out;
#pragma unroll
    for (int i = 0; i < 8; ++i) { out[i] = lo[i]; out[i + 8] = hi[i]; }
    return out;
  };

  v8f acc[4][2] = {};

  issue_tile(0, 0);
  for (int k0 = 0, step = 0; k0 < DM; k0 += 32, ++step) {
    const int  cur  = step & 1;
    const bool more = (k0 + 32) < DM;
    if (more) {
      issue_tile(cur ^ 1, k0 + 32);
      asm volatile("s_wait_asynccnt 0x4" ::: "memory");
    } else {
      asm volatile("s_wait_asynccnt 0x0" ::: "memory");
    }
    __syncthreads();

    v16h af[4], bf[2];
#pragma unroll
    for (int mt = 0; mt < 4; ++mt) af[mt] = frag_lds(sA[cur], wm + mt * 16);
#pragma unroll
    for (int nt = 0; nt < 2; ++nt) bf[nt] = frag_lds(sB[cur], wn + nt * 16);
#pragma unroll
    for (int mt = 0; mt < 4; ++mt)
#pragma unroll
      for (int nt = 0; nt < 2; ++nt)
        acc[mt][nt] = wmma16(af[mt], bf[nt], acc[mt][nt]);

    __syncthreads();
  }

  const int col   = l & 15;
  const int rbase = (l >> 4) << 3;
#pragma unroll
  for (int mt = 0; mt < 4; ++mt)
#pragma unroll
    for (int nt = 0; nt < 2; ++nt) {
      const int n = nB + wn + nt * 16 + col;
#pragma unroll
      for (int v = 0; v < 8; ++v) {
        const int m = mB + wm + mt * 16 + rbase + v;
        Out[(size_t)m * DM + n] = acc[mt][nt][v];
      }
    }
}

// ---------------------------------------------------------------------------
// Causal flash attention. grid = (S/64, B*NH), block = 128 (4 waves).
// One wave owns 16 query rows; kv loop in steps of 32 keys.
// ---------------------------------------------------------------------------
__global__ void flash_attn_kernel(const _Float16* __restrict__ Q,
                                  const _Float16* __restrict__ K,
                                  const _Float16* __restrict__ Vt,
                                  _Float16* __restrict__ A) {
  __shared__ _Float16 Plds[4][16 * LDSTRIDE];   // per-wave 16x32 P tile

  const int bh    = blockIdx.y;
  const int b     = bh >> 4, h = bh & 15;
  const int w     = threadIdx.x >> 5;
  const int l     = threadIdx.x & 31;
  const int col   = l & 15;
  const int rbase = (l >> 4) << 3;
  const int qbase = blockIdx.x * 64 + w * 16;

  const _Float16* Qb = Q + ((size_t)bh * SS + qbase) * HD;
  v16h qa0 = frag_a16(Qb,      HD);   // d = 0..31
  v16h qa1 = frag_a16(Qb + 32, HD);   // d = 32..63

  v8f o[4] = {};
  float mi[8], li[8];
#pragma unroll
  for (int v = 0; v < 8; ++v) { mi[v] = -1e30f; li[v] = 0.0f; }

  const int   qmax  = qbase + 15;
  const float scale = 0.125f;  // 1/sqrt(64)

  for (int j0 = 0; j0 <= qmax; j0 += 32) {
    const _Float16* Kb = K + ((size_t)bh * SS + j0) * HD;
    v16h k00 = frag_a16(Kb,                HD);
    v16h k01 = frag_a16(Kb + 32,           HD);
    v16h k10 = frag_a16(Kb + 16 * HD,      HD);
    v16h k11 = frag_a16(Kb + 16 * HD + 32, HD);

    v8f s0 = {}, s1 = {};
    s0 = wmma16(qa0, k00, s0);
    s0 = wmma16(qa1, k01, s0);
    s1 = wmma16(qa0, k10, s1);
    s1 = wmma16(qa1, k11, s1);

    float fs[8];
#pragma unroll
    for (int v = 0; v < 8; ++v) {
      const int qi = qbase + rbase + v;
      float x0 = (j0 + col      <= qi) ? s0[v] * scale : -1e30f;
      float x1 = (j0 + 16 + col <= qi) ? s1[v] * scale : -1e30f;
      float mx = fmaxf(x0, x1);
#pragma unroll
      for (int msk = 1; msk < 16; msk <<= 1)
        mx = fmaxf(mx, __shfl_xor(mx, msk, 32));   // row max (16 lanes share row)
      const float mnew = fmaxf(mi[v], mx);
      const float f  = __expf(mi[v] - mnew);
      const float e0 = __expf(x0 - mnew);
      const float e1 = __expf(x1 - mnew);
      float rs = e0 + e1;
#pragma unroll
      for (int msk = 1; msk < 16; msk <<= 1)
        rs += __shfl_xor(rs, msk, 32);             // row sum
      li[v] = li[v] * f + rs;
      mi[v] = mnew;
      fs[v] = f;
      Plds[w][(rbase + v) * LDSTRIDE + col]      = (_Float16)e0;
      Plds[w][(rbase + v) * LDSTRIDE + 16 + col] = (_Float16)e1;
    }
#pragma unroll
    for (int t = 0; t < 4; ++t)
#pragma unroll
      for (int v = 0; v < 8; ++v) o[t][v] *= fs[v];

    asm volatile("s_wait_dscnt 0x0" ::: "memory");

    // re-read P in A-fragment layout
    v16h pf;
    {
      const int r  = l & 15;
      const int kk = (l >> 4) << 3;
      const _Float16* q = &Plds[w][r * LDSTRIDE + kk];
      v8h lo = *(const v8h*)(q);
      v8h hi = *(const v8h*)(q + 16);
#pragma unroll
      for (int i = 0; i < 8; ++i) { pf[i] = lo[i]; pf[i + 8] = hi[i]; }
    }

    const _Float16* Vb = Vt + (size_t)bh * HD * SS + j0;
#pragma unroll
    for (int t = 0; t < 4; ++t) {
      v16h vf = frag_a16(Vb + (size_t)(t * 16) * SS, SS);
      o[t] = wmma16(pf, vf, o[t]);
    }
  }

  float inv[8];
#pragma unroll
  for (int v = 0; v < 8; ++v) inv[v] = 1.0f / li[v];
#pragma unroll
  for (int t = 0; t < 4; ++t)
#pragma unroll
    for (int v = 0; v < 8; ++v) {
      const int qi = qbase + rbase + v;
      const int d  = t * 16 + col;
      A[((size_t)(b * SS + qi)) * DM + h * HD + d] =
          (_Float16)(o[t][v] * inv[v]);
    }
}

// ---------------------------------------------------------------------------
extern "C" void kernel_launch(void* const* d_in, const int* in_sizes, int n_in,
                              void* d_out, int out_size, void* d_ws, size_t ws_size,
                              hipStream_t stream) {
  // inputs: [0]=d_model, [1]=num_heads, [2..5]=Wq,Wk,Wv,Wo (fp32), [6]=X (fp32)
  const float* Wq = (const float*)d_in[2];
  const float* Wk = (const float*)d_in[3];
  const float* Wv = (const float*)d_in[4];
  const float* Wo = (const float*)d_in[5];
  const float* X  = (const float*)d_in[6];
  float* Out = (float*)d_out;

  // workspace layout (f16 element offsets), total 24M halves = 48 MB
  _Float16* ws = (_Float16*)d_ws;
  const size_t M1 = (size_t)1024 * 1024;
  _Float16* hX  = ws;              // [4096,1024]
  _Float16* hWq = ws + 4  * M1;
  _Float16* hWk = ws + 5  * M1;
  _Float16* hWv = ws + 6  * M1;
  _Float16* hWo = ws + 7  * M1;
  _Float16* hQ  = ws + 8  * M1;    // [b,h,s,d]
  _Float16* hK  = ws + 12 * M1;    // [b,h,s,d]
  _Float16* hVt = ws + 16 * M1;    // [b,h,d,s]
  _Float16* hA  = ws + 20 * M1;    // [b,s,h*d]

  auto cast = [&](const float* src, _Float16* dst, size_t n) {
    int n4 = (int)(n / 4);
    cast_f16_kernel<<<(n4 + 255) / 256, 256, 0, stream>>>(src, dst, n4);
  };
  cast(X,  hX,  (size_t)MROWS * DM);
  cast(Wq, hWq, M1);
  cast(Wk, hWk, M1);
  cast(Wv, hWv, M1);
  cast(Wo, hWo, M1);

  qkv_gemm_kernel<<<dim3(MROWS / 128, DM / 128, 3), 256, 0, stream>>>(
      hX, hWq, hWk, hWv, hQ, hK, hVt);
  flash_attn_kernel<<<dim3(SS / 64, BB * NH), 128, 0, stream>>>(hQ, hK, hVt, hA);
  o_gemm_kernel<<<dim3(MROWS / 128, DM / 128), 256, 0, stream>>>(hA, hWo, Out);
}
